// MoE_31190052503829
// MI455X (gfx1250) — compile-verified, matrix operand order checked by reading, not run on hardware
//
#include <hip/hip_runtime.h>

typedef unsigned short u16;
typedef unsigned int   u32;
typedef unsigned long long u64;

typedef __bf16 v16bf __attribute__((ext_vector_type(16)));
typedef u16    v8us  __attribute__((ext_vector_type(8)));
typedef u16    v16us __attribute__((ext_vector_type(16)));
typedef float  v8f   __attribute__((ext_vector_type(8)));

#define NTOK   8192
#define CDIM   2048
#define HSH    2816
#define HEXP   1408
#define NEXP   8
#define NSLOT  (NTOK * 2)

// 1 = use CDNA5 GLOBAL_LOAD_ASYNC_TO_LDS_B128 (ASYNCcnt) for GEMM staging
#define USE_ASYNC_LDS 1

// ---------- helpers ----------
__device__ __forceinline__ u16 f2bf(float f) {
    u32 u = __float_as_uint(f);
    u32 r = u + 0x7FFFu + ((u >> 16) & 1u);   // RNE
    return (u16)(r >> 16);
}
__device__ __forceinline__ float bf2f(u16 h) {
    return __uint_as_float(((u32)h) << 16);
}

// 16x32 bf16 WMMA fragment from an LDS row (row-major, K-contig).
// ISA 7.12.2: lanes 0-15 hold K0..7 | K16..23; lanes 16-31 hold K8..15 | K24..31.
__device__ __forceinline__ v16bf load_frag(const u16* row, int kh) {
    v8us lo = *(const v8us*)(row + kh * 8);
    v8us hi = *(const v8us*)(row + 16 + kh * 8);
    v16us u = __builtin_shufflevector(lo, hi, 0,1,2,3,4,5,6,7,8,9,10,11,12,13,14,15);
    union { v16us u; v16bf b; } cv; cv.u = u;
    return cv.b;
}

__device__ __forceinline__ void async_b128(u32 lds_off, u64 gaddr) {
#if USE_ASYNC_LDS
    asm volatile("global_load_async_to_lds_b128 %0, %1, off"
                 :: "v"(lds_off), "v"(gaddr) : "memory");
#endif
}
__device__ __forceinline__ void wait_async() {
#if USE_ASYNC_LDS
    asm volatile("s_wait_asynccnt 0x0" ::: "memory");
#endif
}

// ---------- fp32 -> bf16 cast ----------
__global__ void cast_kernel(const float* __restrict__ in, u16* __restrict__ out, long n) {
    long i = ((long)blockIdx.x * blockDim.x + threadIdx.x) * 8;
    long stride = (long)gridDim.x * blockDim.x * 8;
    for (; i < n; i += stride) {
        float4 a = *(const float4*)(in + i);
        float4 b = *(const float4*)(in + i + 4);
        v8us r;
        r[0]=f2bf(a.x); r[1]=f2bf(a.y); r[2]=f2bf(a.z); r[3]=f2bf(a.w);
        r[4]=f2bf(b.x); r[5]=f2bf(b.y); r[6]=f2bf(b.z); r[7]=f2bf(b.w);
        *(v8us*)(out + i) = r;
    }
}

// ---------- gating: one wave per token ----------
__global__ void gate_kernel(const float* __restrict__ x, const float* __restrict__ gw,
                            const float* __restrict__ gb,
                            int* __restrict__ tidx, float* __restrict__ tw,
                            int N, int C) {
    int wid  = (int)((blockIdx.x * (long)blockDim.x + threadIdx.x) >> 5);
    int lane = threadIdx.x & 31;
    if (wid >= N) return;
    const float* xr = x + (long)wid * C;
    float acc[NEXP];
#pragma unroll
    for (int e = 0; e < NEXP; ++e) acc[e] = 0.f;
    for (int c = lane; c < C; c += 32) {
        float xv = xr[c];
#pragma unroll
        for (int e = 0; e < NEXP; ++e) acc[e] += xv * gw[e * C + c];
    }
#pragma unroll
    for (int e = 0; e < NEXP; ++e)
#pragma unroll
        for (int off = 16; off > 0; off >>= 1)
            acc[e] += __shfl_xor(acc[e], off, 32);
    if (lane == 0) {
        float sc[NEXP], bs[NEXP];
#pragma unroll
        for (int e = 0; e < NEXP; ++e) {
            sc[e] = 1.f / (1.f + expf(-acc[e]));
            bs[e] = sc[e] + gb[e];
        }
        int i0 = 0;
#pragma unroll
        for (int e = 1; e < NEXP; ++e) if (bs[e] > bs[i0]) i0 = e;
        int i1 = -1;
#pragma unroll
        for (int e = 0; e < NEXP; ++e) {
            if (e == i0) continue;
            if (i1 < 0 || bs[e] > bs[i1]) i1 = e;
        }
        float w0 = sc[i0], w1 = sc[i1];
        float inv = 1.f / (w0 + w1);
        tidx[wid * 2 + 0] = i0;
        tidx[wid * 2 + 1] = i1;
        tw[wid * 2 + 0] = w0 * inv;   // ROUTE_SCALE == 1
        tw[wid * 2 + 1] = w1 * inv;
    }
}

// ---------- deterministic per-expert compaction ----------
__global__ void scan_kernel(const int* __restrict__ tidx, int* __restrict__ lists,
                            int* __restrict__ cnt, int N) {
    int e    = threadIdx.x >> 5;
    int lane = threadIdx.x & 31;
    int count = 0;
    for (int base = 0; base < N; base += 32) {
        int n = base + lane;
        int slot = -1;
        if      (tidx[n * 2 + 0] == e) slot = n * 2 + 0;
        else if (tidx[n * 2 + 1] == e) slot = n * 2 + 1;
        unsigned mm = (unsigned)__ballot(slot >= 0);
        if (slot >= 0) {
            int pos = count + __popc(mm & ((1u << lane) - 1u));
            lists[e * N + pos] = slot;
        }
        count += __popc(mm);
    }
    if (lane == 0) cnt[e] = count;
}

// ---------- tiled bf16 WMMA GEMM:  C[M,Nn] = A[M,K] * B[Nn,K]^T ----------
// WG tile 128(M) x 256(N), 8 waves of 64x64, K staged 32/stage, double-buffered LDS,
// staging via GLOBAL_LOAD_ASYNC_TO_LDS_B128 (ASYNCcnt).
template<bool GATHER, bool OUT_BF16>
__global__ void __launch_bounds__(256)
gemm_bf16(const u16* __restrict__ A, const u16* __restrict__ Bw, void* __restrict__ Cout,
          const int* __restrict__ lists, const int* __restrict__ cnt,
          int M, int Nn, int K, int a_shift, long strideB) {
    const int m0 = blockIdx.y * 128;
    const int n0 = blockIdx.x * 256;
    int Me = M;
    const int* lst = nullptr;
    if constexpr (GATHER) {
        const int e = blockIdx.z;
        Me = cnt[e];
        if (m0 >= Me) return;
        lst = lists + (long)e * M;
        Bw += (long)e * strideB;
    }

    __shared__ __align__(16) u16 lA[2][128][40];   // 32 cols + pad (80B rows)
    __shared__ __align__(16) u16 lB[2][256][40];

    const int t   = threadIdx.x;
    const int lr  = t >> 1;          // A load row 0..127
    const int lcA = (t & 1) * 16;    // A col seg (u16): 0 / 16, two b128 each

    long arow;
    if constexpr (GATHER) {
        int i = m0 + lr;
        arow = (i < Me) ? (long)(lst[i] >> a_shift) : 0;  // garbage rows discarded at store
    } else {
        arow = m0 + lr;
    }
    const u16* Aptr = A + arow * K + lcA;
    const u16* Bptr = Bw + (long)(n0 + t) * K;     // one B row per thread (32 cols)

    const u32 laBase = (u32)(size_t)(void*)&lA[0][0][0];
    const u32 lbBase = (u32)(size_t)(void*)&lB[0][0][0];
    const u32 laOff  = laBase + (u32)lr * 80 + (u32)lcA * 2;
    const u32 lbOff  = lbBase + (u32)t * 80;

    struct StageRegs { uint4 a0, a1, b0, b1, b2, b3; };

    auto fetch = [&](int kk, int buf) -> StageRegs {
        StageRegs r;
#if USE_ASYNC_LDS
        u32 la = laOff + (u32)buf * (128 * 80);
        u32 lb = lbOff + (u32)buf * (256 * 80);
        u64 ga = (u64)(Aptr + kk);
        u64 gb = (u64)(Bptr + kk);
        async_b128(la,      ga);
        async_b128(la + 16, ga + 16);
        async_b128(lb,      gb);
        async_b128(lb + 16, gb + 16);
        async_b128(lb + 32, gb + 32);
        async_b128(lb + 48, gb + 48);
        (void)r;
#else
        r.a0 = *(const uint4*)(Aptr + kk);
        r.a1 = *(const uint4*)(Aptr + kk + 8);
        r.b0 = *(const uint4*)(Bptr + kk);
        r.b1 = *(const uint4*)(Bptr + kk + 8);
        r.b2 = *(const uint4*)(Bptr + kk + 16);
        r.b3 = *(const uint4*)(Bptr + kk + 24);
#endif
        return r;
    };
    auto commit = [&](int buf, const StageRegs& r) {
#if USE_ASYNC_LDS
        (void)buf; (void)r;
        wait_async();
#else
        *(uint4*)&lA[buf][lr][lcA]      = r.a0;
        *(uint4*)&lA[buf][lr][lcA + 8]  = r.a1;
        *(uint4*)&lB[buf][t][0]         = r.b0;
        *(uint4*)&lB[buf][t][8]         = r.b1;
        *(uint4*)&lB[buf][t][16]        = r.b2;
        *(uint4*)&lB[buf][t][24]        = r.b3;
#endif
    };

    const int lane = t & 31;
    const int wave = t >> 5;
    const int wm   = (wave & 1) * 64;    // 2 M-wave groups
    const int wn   = (wave >> 1) * 64;   // 4 N-wave groups
    const int rsel = lane & 15;
    const int kh   = lane >> 4;

    v8f acc[4][4] = {};

    const int nk = K >> 5;   // K / 32 stages
    {
        StageRegs s0 = fetch(0, 0);
        commit(0, s0);
    }
    __syncthreads();

    for (int kk = 0; kk < nk; ++kk) {
        const int buf = kk & 1;
        StageRegs nxt{};
        const bool have = (kk + 1) < nk;
        if (have) nxt = fetch((kk + 1) << 5, buf ^ 1);

        v16bf afr[4], bfr[4];
#pragma unroll
        for (int mi = 0; mi < 4; ++mi)
            afr[mi] = load_frag(&lA[buf][wm + mi * 16 + rsel][0], kh);
#pragma unroll
        for (int ni = 0; ni < 4; ++ni)
            bfr[ni] = load_frag(&lB[buf][wn + ni * 16 + rsel][0], kh);
#pragma unroll
        for (int mi = 0; mi < 4; ++mi)
#pragma unroll
            for (int ni = 0; ni < 4; ++ni)
                acc[mi][ni] = __builtin_amdgcn_wmma_f32_16x16x32_bf16(
                    false, afr[mi], false, bfr[ni], (short)0, acc[mi][ni], false, false);

        if (have) {
            __syncthreads();            // all waves done reading buf^1 (stage kk-1)
            commit(buf ^ 1, nxt);
            __syncthreads();            // stage kk+1 data visible
        }
    }

    // epilogue: acc VGPR r -> (M = r + kh*8, N = rsel) within each 16x16 tile
    const int rowhalf = kh * 8;
#pragma unroll
    for (int mi = 0; mi < 4; ++mi) {
#pragma unroll
        for (int r = 0; r < 8; ++r) {
            int i = m0 + wm + mi * 16 + rowhalf + r;
            long crow;
            if constexpr (GATHER) {
                if (i >= Me) continue;
                crow = lst[i];
            } else {
                if (i >= M) continue;
                crow = i;
            }
#pragma unroll
            for (int ni = 0; ni < 4; ++ni) {
                int nn = n0 + wn + ni * 16 + rsel;
                float v = acc[mi][ni][r];
                if constexpr (OUT_BF16)
                    ((u16*)Cout)[crow * Nn + nn] = f2bf(v);
                else
                    ((float*)Cout)[crow * Nn + nn] = v;
            }
        }
    }
}

// ---------- SwiGLU: out[r,j] = silu(in[r,off_g+j]) * in[r,off_v+j] ----------
__global__ void swiglu_kernel(const u16* __restrict__ in, u16* __restrict__ out,
                              int rows, int H, int stride, int off_val, int off_gate) {
    int r = blockIdx.y;
    int j = blockIdx.x * blockDim.x + threadIdx.x;
    if (r >= rows || j >= H) return;
    float g = bf2f(in[(long)r * stride + off_gate + j]);
    float v = bf2f(in[(long)r * stride + off_val + j]);
    float s = g / (1.f + expf(-g));
    out[(long)r * H + j] = f2bf(s * v);
}

// ---------- deterministic combine ----------
__global__ void combine_kernel(float* __restrict__ out, const float* __restrict__ yc,
                               const float* __restrict__ w, int C) {
    int n = blockIdx.y;
    int c = blockIdx.x * blockDim.x + threadIdx.x;
    if (c >= C) return;
    float w0 = w[n * 2 + 0], w1 = w[n * 2 + 1];
    long o = (long)n * C + c;
    out[o] += w0 * yc[(long)(n * 2 + 0) * C + c] + w1 * yc[(long)(n * 2 + 1) * C + c];
}

// ---------- host launch ----------
extern "C" void kernel_launch(void* const* d_in, const int* in_sizes, int n_in,
                              void* d_out, int out_size, void* d_ws, size_t ws_size,
                              hipStream_t stream) {
    (void)in_sizes; (void)n_in; (void)out_size; (void)ws_size;
    const int N = NTOK, C = CDIM, HS = HSH, H = HEXP, E = NEXP, S = NSLOT;

    const float* x     = (const float*)d_in[0];
    const float* gw    = (const float*)d_in[1];
    const float* gb    = (const float*)d_in[2];
    const float* sgw_f = (const float*)d_in[3];
    const float* sdw_f = (const float*)d_in[4];
    const float* upw_f = (const float*)d_in[5];
    const float* dnw_f = (const float*)d_in[6];
    float* out = (float*)d_out;

    char* ws = (char*)d_ws;
    size_t off = 0;
    auto alloc = [&](size_t bytes) -> void* {
        off = (off + 255) & ~(size_t)255;
        void* p = ws + off;
        off += bytes;
        return p;
    };
    u16* xb   = (u16*)alloc((size_t)N * C * 2);
    u16* sgw  = (u16*)alloc((size_t)2 * HS * C * 2);
    u16* sdw  = (u16*)alloc((size_t)C * HS * 2);
    u16* upw  = (u16*)alloc((size_t)E * 2 * H * C * 2);
    u16* dnw  = (u16*)alloc((size_t)E * C * H * 2);
    u16* yg   = (u16*)alloc((size_t)N * 2 * HS * 2);  // N*2HS == S*2H (aliased)
    u16* gu   = yg;
    u16* hshd = (u16*)alloc((size_t)N * HS * 2);      // N*HS == S*H (aliased)
    u16* hcmp = hshd;
    float* yc = (float*)alloc((size_t)S * C * 4);
    int*   tidx  = (int*)alloc((size_t)N * 2 * 4);
    float* tw    = (float*)alloc((size_t)N * 2 * 4);
    int*   lists = (int*)alloc((size_t)E * N * 4);
    int*   cnt   = (int*)alloc((size_t)E * 4);

    // 1) casts to bf16
    cast_kernel<<<2048, 256, 0, stream>>>(x,     xb,  (long)N * C);
    cast_kernel<<<2048, 256, 0, stream>>>(sgw_f, sgw, (long)2 * HS * C);
    cast_kernel<<<2048, 256, 0, stream>>>(sdw_f, sdw, (long)C * HS);
    cast_kernel<<<2048, 256, 0, stream>>>(upw_f, upw, (long)E * 2 * H * C);
    cast_kernel<<<2048, 256, 0, stream>>>(dnw_f, dnw, (long)E * C * H);

    // 2) gating + deterministic compaction
    gate_kernel<<<(N * 32) / 256, 256, 0, stream>>>(x, gw, gb, tidx, tw, N, C);
    scan_kernel<<<1, 256, 0, stream>>>(tidx, lists, cnt, N);

    // 3) shared expert
    gemm_bf16<false, true><<<dim3(2 * HS / 256, N / 128, 1), 256, 0, stream>>>(
        xb, sgw, yg, nullptr, nullptr, N, 2 * HS, C, 0, 0);
    swiglu_kernel<<<dim3(HS / 128, N), 128, 0, stream>>>(yg, hshd, N, HS, 2 * HS, 0, HS);
    gemm_bf16<false, false><<<dim3(C / 256, N / 128, 1), 256, 0, stream>>>(
        hshd, sdw, out, nullptr, nullptr, N, C, HS, 0, 0);

    // 4) routed experts (grouped via slot lists)
    gemm_bf16<true, true><<<dim3(2 * H / 256, N / 128, E), 256, 0, stream>>>(
        xb, upw, gu, lists, cnt, N, 2 * H, C, /*a_shift=*/1, (long)2 * H * C);
    swiglu_kernel<<<dim3(H / 128, S), 128, 0, stream>>>(gu, hcmp, S, H, 2 * H, H, 0);
    gemm_bf16<true, false><<<dim3(C / 256, N / 128, E), 256, 0, stream>>>(
        hcmp, dnw, yc, lists, cnt, N, C, H, /*a_shift=*/0, (long)C * H);

    // 5) weighted combine into d_out
    combine_kernel<<<dim3(C / 256, N), 256, 0, stream>>>(out, yc, tw, C);
}
